// PointNetSetAbstraction_74062416052697
// MI455X (gfx1250) — matrix operand order repaired and checked by reading, compiled, not run on hardware
//
#include <hip/hip_runtime.h>

// ---------------------------------------------------------------------------
// PointNet++ Set Abstraction for MI455X (gfx1250, wave32, WMMA).
// B=8, N=4096, NPOINT=1024, NSAMPLE=32, IN_CH=64, MLP=[64,64,128].
// Notes:
//  * conv bias b{i} cancels exactly inside global batch-norm -> dropped.
//  * GEMMs run as bf16 WMMA (v_wmma_f32_16x16x32_bf16) with f32 accum;
//    N/K are template params so fragments stay in static VGPRs.
//  * BN statistics (per-channel sum / sumsq over all 262144 samples) are
//    computed in the GEMM epilogue from the accumulator registers
//    (deterministic fixed-order reduction, no FP atomics), then finished by
//    a small partial-reduce kernel -> saves a full re-read of Y per layer.
//  * BN+ReLU fused into the bf16 repack (layers 0,1) / maxpool (layer 2).
// ---------------------------------------------------------------------------

#define BB 8
#define NN 4096
#define NPT 1024
#define NSMP 32
#define INCH 64
#define R2 (0.2f * 0.2f)
#define EPSV 1e-5f

#define MROWS (BB * NPT * NSMP)       // 262144 samples through the MLP
#define K0PAD 96                      // 67 -> pad to 3*32 for bf16 WMMA K-steps
#define CNTF 262144.0f                // BN reduction count
#define GEMM_BLOCKS (MROWS / 16 / 8)  // 2048 (8 waves/block, 16 rows/wave)

typedef __attribute__((ext_vector_type(16))) __bf16 v16bf;
typedef __attribute__((ext_vector_type(8)))  __bf16 bf16x8;
typedef __attribute__((ext_vector_type(2)))  __bf16 bf16x2;
typedef __attribute__((ext_vector_type(8)))  float  v8f;

union ABFrag { v16bf v; bf16x8 h[2]; };

static __device__ __forceinline__ __bf16 tobf(float f) { return (__bf16)f; }

// ---------------------------------------------------------------------------
// 1) Farthest point sampling: one block per batch. xyz in LDS, distance in
//    registers (4 pts/thread), shuffle+LDS argmax with lowest-index ties.
//    Two barriers per step (double-buffered result slot).
//    Writes new_xyz directly into d_out[0 .. B*NPT*3).
// ---------------------------------------------------------------------------
__global__ __launch_bounds__(1024) void fps_kernel(const float* __restrict__ xyz,
                                                   float* __restrict__ out_xyz) {
  const int b = blockIdx.x;
  const int tid = threadIdx.x;
  const int lane = tid & 31;
  const int wid = tid >> 5;

  __shared__ float sx[NN], sy[NN], sz[NN];
  __shared__ float rv[32];
  __shared__ int   ri[32];
  __shared__ int   rsel[2];

  float px[4], py[4], pz[4], dist[4];
#pragma unroll
  for (int t = 0; t < 4; ++t) {
    int i = tid + t * 1024;
    const float* p = xyz + ((size_t)b * NN + i) * 3;
    px[t] = p[0]; py[t] = p[1]; pz[t] = p[2];
    sx[i] = px[t]; sy[i] = py[t]; sz[i] = pz[t];
    dist[t] = 1e10f;
  }
  __syncthreads();

  int cur = 0;
  for (int it = 0; it < NPT; ++it) {
    if (tid == 0) {
      float* o = out_xyz + ((size_t)b * NPT + it) * 3;
      o[0] = sx[cur]; o[1] = sy[cur]; o[2] = sz[cur];
    }
    const float cx = sx[cur], cy = sy[cur], cz = sz[cur];

    float bv = -1.0f; int bi = 0x7fffffff;
#pragma unroll
    for (int t = 0; t < 4; ++t) {
      float dx = px[t] - cx, dy = py[t] - cy, dz = pz[t] - cz;
      float dd = dx * dx + dy * dy + dz * dz;
      dist[t] = fminf(dist[t], dd);
      int i = tid + t * 1024;
      if (dist[t] > bv || (dist[t] == bv && i < bi)) { bv = dist[t]; bi = i; }
    }
    // wave32 reduce (prefer larger value, then smaller index)
#pragma unroll
    for (int off = 16; off > 0; off >>= 1) {
      float ov = __shfl_down(bv, off);
      int   oi = __shfl_down(bi, off);
      if (ov > bv || (ov == bv && oi < bi)) { bv = ov; bi = oi; }
    }
    if (lane == 0) { rv[wid] = bv; ri[wid] = bi; }
    __syncthreads();                 // (A) partials visible
    if (wid == 0) {
      bv = rv[lane]; bi = ri[lane];
#pragma unroll
      for (int off = 16; off > 0; off >>= 1) {
        float ov = __shfl_down(bv, off);
        int   oi = __shfl_down(bi, off);
        if (ov > bv || (ov == bv && oi < bi)) { bv = ov; bi = oi; }
      }
      if (lane == 0) rsel[it & 1] = bi;
    }
    __syncthreads();                 // (B) result visible
    cur = rsel[it & 1];              // slot it&1 not rewritten until it+2,
                                     // which is gated behind barrier A of it+2
  }
}

// ---------------------------------------------------------------------------
// 2) Ball query: one wave per (b,s). Ordered chunk scan; ballot + popc prefix
//    collects the 32 smallest in-radius indices; pad with first (always >=1
//    since each center is itself a dataset point).
// ---------------------------------------------------------------------------
__global__ __launch_bounds__(32) void qb_kernel(const float* __restrict__ xyz,
                                                const float* __restrict__ newxyz,
                                                int* __restrict__ gidx) {
  const int bs = blockIdx.x;          // b*NPT + s
  const int b = bs >> 10;
  const int lane = threadIdx.x;
  __shared__ int sidx[NSMP];

  const float qx = newxyz[(size_t)bs * 3 + 0];
  const float qy = newxyz[(size_t)bs * 3 + 1];
  const float qz = newxyz[(size_t)bs * 3 + 2];

  int cnt = 0;
  for (int c = 0; c < NN / 32 && cnt < NSMP; ++c) {
    int i = c * 32 + lane;
    const float* p = xyz + ((size_t)b * NN + i) * 3;
    float dx = p[0] - qx, dy = p[1] - qy, dz = p[2] - qz;
    bool in = (dx * dx + dy * dy + dz * dz) <= R2;
    unsigned m32 = (unsigned)__ballot(in);
    int rank = __popc(m32 & ((1u << lane) - 1u));
    if (in) {
      int slot = cnt + rank;
      if (slot < NSMP) sidx[slot] = i;
    }
    cnt += __popc(m32);
  }
  __syncthreads();
  int take = cnt < NSMP ? cnt : NSMP;
  int v = (take > 0) ? ((lane < take) ? sidx[lane] : sidx[0]) : 0;
  gidx[(size_t)bs * NSMP + lane] = v;
}

// ---------------------------------------------------------------------------
// 3a) Weight prep: f32 -> bf16, channel-permuted (points first, xyz last),
//     K padded with zeros. wB0: 64x96, wB1: 64x64, wB2: 128x64.
// ---------------------------------------------------------------------------
__global__ void prep_weights(const float* __restrict__ w0,
                             const float* __restrict__ w1,
                             const float* __restrict__ w2,
                             __bf16* __restrict__ wB0,
                             __bf16* __restrict__ wB1,
                             __bf16* __restrict__ wB2) {
  int id = blockIdx.x * blockDim.x + threadIdx.x;
  if (id < 64 * K0PAD) {
    int o = id / K0PAD, j = id % K0PAD;
    float v = 0.0f;
    if (j < 64)       v = w0[o * 67 + 3 + j];    // point features
    else if (j < 67)  v = w0[o * 67 + (j - 64)]; // relative xyz
    wB0[id] = tobf(v);
  } else if (id < 64 * K0PAD + 64 * 64) {
    int k = id - 64 * K0PAD;
    wB1[k] = tobf(w1[k]);
  } else if (id < 64 * K0PAD + 64 * 64 + 128 * 64) {
    int k = id - 64 * K0PAD - 64 * 64;
    wB2[k] = tobf(w2[k]);
  }
}

// ---------------------------------------------------------------------------
// 3b) Gather + concat -> bf16 activation rows (M x 96), channels:
//     [0..63]=points[gidx], [64..66]=xyz[gidx]-new_xyz, [67..95]=0.
// ---------------------------------------------------------------------------
__global__ __launch_bounds__(256) void build_xin(const float* __restrict__ xyz,
                                                 const float* __restrict__ pts,
                                                 const float* __restrict__ newxyz,
                                                 const int* __restrict__ gidx,
                                                 __bf16* __restrict__ X) {
  int m = blockIdx.x * blockDim.x + threadIdx.x;   // 0..MROWS
  if (m >= MROWS) return;
  int b = m >> 15;
  int gi = gidx[m];
  const float* p = pts + ((size_t)b * NN + gi) * INCH;
  size_t ro = (size_t)m * K0PAD;
#pragma unroll
  for (int j = 0; j < 32; ++j) {
    bf16x2 pr = { tobf(p[2 * j]), tobf(p[2 * j + 1]) };
    *(bf16x2*)(X + ro + 2 * j) = pr;
  }
  const float* g = xyz + ((size_t)b * NN + gi) * 3;
  const float* nx = newxyz + (size_t)(m >> 5) * 3;
  bf16x2 a = { tobf(g[0] - nx[0]), tobf(g[1] - nx[1]) };
  bf16x2 c = { tobf(g[2] - nx[2]), tobf(0.0f) };
  *(bf16x2*)(X + ro + 64) = a;
  *(bf16x2*)(X + ro + 66) = c;
  bf16x2 z = { tobf(0.0f), tobf(0.0f) };
#pragma unroll
  for (int j = 68; j < K0PAD; j += 2) *(bf16x2*)(X + ro + j) = z;
}

// ---------------------------------------------------------------------------
// 4) bf16 WMMA GEMM + fused BN-stat partials.
//    Y[M,N] = X[M,K] * W[N,K]^T, f32 accumulate. 256 threads = 8 waves;
//    W staged in LDS; each wave owns one 16-row tile, A fragments reused
//    across all N-tiles. Epilogue reduces acc / acc^2 per column:
//    8 serial adds + shfl_xor(16) per tile, per-wave LDS slots, fixed-order
//    cross-wave combine -> one deterministic partial per (block, channel).
// ---------------------------------------------------------------------------
template <int N, int K>
__global__ __launch_bounds__(256) void gemm_wmma(const __bf16* __restrict__ X,
                                                 const __bf16* __restrict__ Wb,
                                                 float* __restrict__ Y,
                                                 float* __restrict__ Psum,
                                                 float* __restrict__ Psq) {
  constexpr int KT = K / 32;
  constexpr int NT = N / 16;
  __shared__ __bf16 sW[N * K];
  __shared__ float sS[8 * N];
  __shared__ float sQ[8 * N];

  const int lane = threadIdx.x & 31;
  const int wid  = threadIdx.x >> 5;

#pragma unroll 4
  for (int i = threadIdx.x; i < N * K; i += 256) sW[i] = Wb[i];
  __syncthreads();

  const int mtile = blockIdx.x * 8 + wid;

  // A fragments: lane<16 holds row M=lane, K {kb..kb+7, kb+16..kb+23}, kb=0;
  //              lane>=16 same row set with kb=8.  (ISA 7.12.2, 16-bit A)
  const int arow = (mtile << 4) + (lane & 15);
  const int kb = (lane >> 4) << 3;
  v16bf afr[KT];
#pragma unroll
  for (int kt = 0; kt < KT; ++kt) {
    const __bf16* ap = X + (size_t)arow * K + (kt << 5) + kb;
    ABFrag u;
    u.h[0] = *(const bf16x8*)(ap);
    u.h[1] = *(const bf16x8*)(ap + 16);
    afr[kt] = u.v;
  }

  // B fragments: lane = N-col (lane&15), K {kb..kb+7, kb+16..kb+23}, kb=(lane>>4)*8 (matches A).
  const int bn = lane & 15;
  const int bk = (lane >> 4) << 3;

#pragma unroll
  for (int nt = 0; nt < NT; ++nt) {
    const int orow = (nt << 4) + bn;
    v16bf bfr[KT];
#pragma unroll
    for (int kt = 0; kt < KT; ++kt) {
      const __bf16* bp = &sW[orow * K + (kt << 5) + bk];
      ABFrag ub;
      ub.h[0] = *(const bf16x8*)(bp);
      ub.h[1] = *(const bf16x8*)(bp + 16);
      bfr[kt] = ub.v;
    }
    v8f acc = {};
#pragma unroll
    for (int kt = 0; kt < KT; ++kt) {
      acc = __builtin_amdgcn_wmma_f32_16x16x32_bf16(
          false, afr[kt], false, bfr[kt], (short)0, acc, false, false);
    }
    // D layout: vgpr r -> row (lane>>4)*8 + r, col = lane&15 (ISA 7.12.2)
    const int srow = (mtile << 4) + ((lane >> 4) << 3);
    const int col  = (nt << 4) + bn;
    float* yp = Y + (size_t)srow * N + col;
    float s8 = 0.0f, q8 = 0.0f;
#pragma unroll
    for (int r = 0; r < 8; ++r) {
      yp[(size_t)r * N] = acc[r];
      s8 += acc[r];
      q8 += acc[r] * acc[r];
    }
    // combine the two 8-row halves of this column (lanes l and l^16)
    s8 += __shfl_xor(s8, 16);
    q8 += __shfl_xor(q8, 16);
    if (lane < 16) {
      sS[wid * N + col] = s8;
      sQ[wid * N + col] = q8;
    }
  }
  __syncthreads();
  // fixed-order cross-wave combine -> per-block partials (deterministic)
  for (int c = threadIdx.x; c < N; c += 256) {
    float s = 0.0f, q = 0.0f;
#pragma unroll
    for (int w8 = 0; w8 < 8; ++w8) { s += sS[w8 * N + c]; q += sQ[w8 * N + c]; }
    Psum[(size_t)blockIdx.x * N + c] = s;
    Psq[(size_t)blockIdx.x * N + c] = q;
  }
}

// ---------------------------------------------------------------------------
// 5) Finish BN stats: reduce GEMM_BLOCKS partials per channel (fixed tree).
// ---------------------------------------------------------------------------
__global__ __launch_bounds__(256) void stats_final(const float* __restrict__ Psum,
                                                   const float* __restrict__ Psq,
                                                   float* __restrict__ stats,
                                                   int N) {
  __shared__ float ssum[256], ssq[256];
  const int ch = blockIdx.x;
  float s = 0.0f, q = 0.0f;
  for (int i = threadIdx.x; i < GEMM_BLOCKS; i += 256) {
    s += Psum[(size_t)i * N + ch];
    q += Psq[(size_t)i * N + ch];
  }
  ssum[threadIdx.x] = s; ssq[threadIdx.x] = q;
  __syncthreads();
  for (int off = 128; off > 0; off >>= 1) {
    if (threadIdx.x < off) {
      ssum[threadIdx.x] += ssum[threadIdx.x + off];
      ssq[threadIdx.x]  += ssq[threadIdx.x + off];
    }
    __syncthreads();
  }
  if (threadIdx.x == 0) { stats[ch] = ssum[0]; stats[128 + ch] = ssq[0]; }
}

// ---------------------------------------------------------------------------
// 6) BN + ReLU -> bf16 repack for the next GEMM (layers 0,1; N=64).
// ---------------------------------------------------------------------------
__global__ __launch_bounds__(256) void bnrelu_pack(const float* __restrict__ Y,
                                                   const float* __restrict__ stats,
                                                   const float* __restrict__ g,
                                                   const float* __restrict__ bt,
                                                   __bf16* __restrict__ X,
                                                   int N) {
  size_t idx = (size_t)blockIdx.x * blockDim.x + threadIdx.x;
  size_t e = idx * 2;
  if (e >= (size_t)MROWS * N) return;
  int ch = (int)(e % N);   // even
#pragma unroll
  for (int u = 0; u < 2; ++u) {
    int c = ch + u;
    float mu = stats[c] / CNTF;
    float var = stats[128 + c] / CNTF - mu * mu;
    float rs = rsqrtf(var + EPSV);
    float sc = g[c] * rs;
    float sh = bt[c] - mu * sc;
    float v = fmaxf(fmaf(Y[e + u], sc, sh), 0.0f);
    X[e + u] = tobf(v);
  }
}

// ---------------------------------------------------------------------------
// 7) Final layer: BN + ReLU + max over NSMP, write new_points to d_out.
// ---------------------------------------------------------------------------
__global__ __launch_bounds__(128) void bnrelu_max(const float* __restrict__ Y,
                                                  const float* __restrict__ stats,
                                                  const float* __restrict__ g,
                                                  const float* __restrict__ bt,
                                                  float* __restrict__ out) {
  const int bs = blockIdx.x;       // b*NPT + s
  const int c = threadIdx.x;       // 0..127
  float mu = stats[c] / CNTF;
  float var = stats[128 + c] / CNTF - mu * mu;
  float rs = rsqrtf(var + EPSV);
  float sc = g[c] * rs;
  float sh = bt[c] - mu * sc;
  size_t base = (size_t)bs * NSMP;
  float best = 0.0f;               // ReLU outputs are >= 0
  for (int k = 0; k < NSMP; ++k) {
    float v = fmaxf(fmaf(Y[(base + k) * 128 + c], sc, sh), 0.0f);
    best = fmaxf(best, v);
  }
  out[(size_t)BB * NPT * 3 + (size_t)bs * 128 + c] = best;
}

// ---------------------------------------------------------------------------
extern "C" void kernel_launch(void* const* d_in, const int* in_sizes, int n_in,
                              void* d_out, int out_size, void* d_ws, size_t ws_size,
                              hipStream_t stream) {
  const float* xyz = (const float*)d_in[0];
  const float* pts = (const float*)d_in[1];
  const float* w0  = (const float*)d_in[2];
  const float* g0  = (const float*)d_in[4];
  const float* bt0 = (const float*)d_in[5];
  const float* w1  = (const float*)d_in[6];
  const float* g1  = (const float*)d_in[8];
  const float* bt1 = (const float*)d_in[9];
  const float* w2  = (const float*)d_in[10];
  const float* g2  = (const float*)d_in[12];
  const float* bt2 = (const float*)d_in[13];
  float* out = (float*)d_out;

  // workspace layout (256B-aligned regions)
  char* w = (char*)d_ws;
  int*    groupIdx = (int*)w;                      // 1,048,576 B
  __bf16* wB0 = (__bf16*)(w + 1048576);            // 64*96 bf16
  __bf16* wB1 = wB0 + 64 * K0PAD;                  // 64*64 bf16
  __bf16* wB2 = wB1 + 64 * 64;                     // 128*64 bf16 (ends 1,085,440)
  float*  stats = (float*)(w + 1085440);           // 256 f32   (ends 1,086,464)
  float*  Psum  = (float*)(w + 1086464);           // 2048*128 f32 (1 MB)
  float*  Psq   = (float*)(w + 2135040);           // 2048*128 f32 (1 MB)
  __bf16* Xbuf  = (__bf16*)(w + 3183616);          // M*96 bf16 (48 MB)
  float*  Ybuf  = (float*)(w + 53515264);          // M*128 f32 (128 MB)

  // 1) FPS -> new_xyz into d_out
  fps_kernel<<<BB, 1024, 0, stream>>>(xyz, out);
  // 2) ball query
  qb_kernel<<<BB * NPT, 32, 0, stream>>>(xyz, out, groupIdx);
  // 3) weights + layer-0 input
  prep_weights<<<72, 256, 0, stream>>>(w0, w1, w2, wB0, wB1, wB2);
  build_xin<<<MROWS / 256, 256, 0, stream>>>(xyz, pts, out, groupIdx, Xbuf);

  // layer 0: 96 -> 64
  gemm_wmma<64, K0PAD><<<GEMM_BLOCKS, 256, 0, stream>>>(Xbuf, wB0, Ybuf, Psum, Psq);
  stats_final<<<64, 256, 0, stream>>>(Psum, Psq, stats, 64);
  bnrelu_pack<<<(MROWS * 64 / 2 + 255) / 256, 256, 0, stream>>>(Ybuf, stats, g0, bt0, Xbuf, 64);

  // layer 1: 64 -> 64
  gemm_wmma<64, 64><<<GEMM_BLOCKS, 256, 0, stream>>>(Xbuf, wB1, Ybuf, Psum, Psq);
  stats_final<<<64, 256, 0, stream>>>(Psum, Psq, stats, 64);
  bnrelu_pack<<<(MROWS * 64 / 2 + 255) / 256, 256, 0, stream>>>(Ybuf, stats, g1, bt1, Xbuf, 64);

  // layer 2: 64 -> 128, then BN+ReLU+max -> d_out
  gemm_wmma<128, 64><<<GEMM_BLOCKS, 256, 0, stream>>>(Xbuf, wB2, Ybuf, Psum, Psq);
  stats_final<<<128, 256, 0, stream>>>(Psum, Psq, stats, 128);
  bnrelu_max<<<BB * NPT, 128, 0, stream>>>(Ybuf, stats, g2, bt2, out);
}
